// FastAttention_56367150793422
// MI455X (gfx1250) — matrix operand order repaired and checked by reading, not compiled, not run on hardware
//
#include <hip/hip_runtime.h>
#include <hip/hip_bf16.h>

// ---------------------------------------------------------------------------
// Problem constants (from reference): B=2, T=2048, C=1024, H=16, D=M=64
// ---------------------------------------------------------------------------
#define BB   2
#define TT   2048
#define CC   1024
#define HH   16
#define DD   64
#define MM   64
#define LC   64      // scan chunk length (timesteps per chunk)
#define ST   72      // LDS row stride (elements) for 64-wide bf16 tiles (16B aligned)

typedef __bf16 bf16_t;
typedef __attribute__((ext_vector_type(16))) __bf16 v16bf;
typedef __attribute__((ext_vector_type(8)))  float  v8f;

__device__ inline v8f wmma_bf16(v16bf a, v16bf b, v8f c) {
  // D = A(16x32) * B(32x16) + C, f32 accumulate
  return __builtin_amdgcn_wmma_f32_16x16x32_bf16(
      /*neg_a=*/false, a, /*neg_b=*/false, b,
      /*c_mod=*/(short)0, c, /*reuse_a=*/false, /*reuse_b=*/false);
}

// A-matrix fragment (16x32 bf16, row-major LDS tile base[row][k], stride ld):
// lane l: row m = l&15; lanes 0-15 hold K={0..7,16..23}, lanes 16-31 K={8..15,24..31}
__device__ inline v16bf load_frag_a(const bf16_t* base, int row0, int k0, int ld) {
  const int lane = threadIdx.x & 31;
  const int m  = lane & 15;
  const int kh = (lane >> 4) * 8;
  const bf16_t* p = base + (row0 + m) * ld + k0 + kh;
  union { v16bf v; uint4 q[2]; } u;
  u.q[0] = *(const uint4*)(p);        // K = kh .. kh+7
  u.q[1] = *(const uint4*)(p + 16);   // K = kh+16 .. kh+23
  return u.v;
}

// B-matrix fragment (32x16 bf16, LDS tile stored column-major base[n][k], stride ld):
// lane l: col n = l&15; lanes 0-15 hold K={0..7,16..23}, lanes 16-31 K={8..15,24..31}
__device__ inline v16bf load_frag_b(const bf16_t* base, int n0, int k0, int ld) {
  const int lane = threadIdx.x & 31;
  const int n  = lane & 15;
  const int kb = (lane >> 4) * 8;
  const bf16_t* p = base + (n0 + n) * ld + k0 + kb;
  union { v16bf v; uint4 q[2]; } u;
  u.q[0] = *(const uint4*)(p);        // K = kb .. kb+7
  u.q[1] = *(const uint4*)(p + 16);   // K = kb+16 .. kb+23
  return u.v;
}

__device__ inline void store_bf16x4(bf16_t* dst, float4 f) {
  union { bf16_t h[4]; uint2 u; } t;
  t.h[0] = (bf16_t)f.x; t.h[1] = (bf16_t)f.y;
  t.h[2] = (bf16_t)f.z; t.h[3] = (bf16_t)f.w;
  *(uint2*)dst = t.u;
}

// ---------------------------------------------------------------------------
// GEMM with bias: Cmat[M][N] = A[M][K] @ W[N][K]^T + bias[N]   (f32 in/out)
// Block tile 128x128, K-step 32, 256 threads = 8 waves (4 row-strips x 2 col-strips),
// each wave computes 32x64 = 2x4 WMMA tiles.
// Double-buffered LDS: next K-tile's global loads are issued before the barrier
// and stay in flight across the WMMA block (one barrier per iteration).
// ---------------------------------------------------------------------------
#define BM 128
#define BN 128
#define BK 32
#define SA 40
#define SB 40

__global__ __launch_bounds__(256) void gemm_bias_wmma(
    const float* __restrict__ A, const float* __restrict__ W,
    const float* __restrict__ bias, float* __restrict__ Cmat,
    int M, int N, int K)
{
  __shared__ bf16_t As[2][BM * SA];   // row-major [row][k]
  __shared__ bf16_t Bs[2][BN * SB];   // col-major [n][k]  (== W tile rows)

  const int tid  = threadIdx.x;
  const int wave = tid >> 5;
  const int lane = tid & 31;
  const int ln = lane & 15, lh = lane >> 4;
  const int wr = wave & 3;        // row strip 0..3 (32 rows each)
  const int wc = wave >> 2;       // col strip 0..1 (64 cols each)
  const int rowBase = blockIdx.x * BM;
  const int colBase = blockIdx.y * BN;

  const v8f vzero = {0.f, 0.f, 0.f, 0.f, 0.f, 0.f, 0.f, 0.f};
  v8f acc[2][4];
#pragma unroll
  for (int mi = 0; mi < 2; ++mi)
#pragma unroll
    for (int ni = 0; ni < 4; ++ni) acc[mi][ni] = vzero;

  const int tk = tid & 7;    // float4 slot in 32-wide K
  const int tr = tid >> 3;   // 0..31

  const float* aptr = A + (size_t)(rowBase + tr) * K + tk * 4;  // +32*p rows
  const float* wptr = W + (size_t)(colBase + tr) * K + tk * 4;

  float4 ra[4], rb[4];

  // ---- prologue: stage K-tile 0 into buffer 0 ----
#pragma unroll
  for (int p = 0; p < 4; ++p) {
    ra[p] = *(const float4*)(aptr + (size_t)(p * 32) * K);
    rb[p] = *(const float4*)(wptr + (size_t)(p * 32) * K);
  }
#pragma unroll
  for (int p = 0; p < 4; ++p) {
    store_bf16x4(&As[0][(tr + p * 32) * SA + tk * 4], ra[p]);
    store_bf16x4(&Bs[0][(tr + p * 32) * SB + tk * 4], rb[p]);
  }

  int cur = 0;
  for (int kb = 0; kb < K; kb += BK) {
    const bool has_next = (kb + BK) < K;
    if (has_next) {  // issue next tile's loads; they stay in flight across the WMMAs
#pragma unroll
      for (int p = 0; p < 4; ++p) {
        ra[p] = *(const float4*)(aptr + (size_t)(p * 32) * K + kb + BK);
        rb[p] = *(const float4*)(wptr + (size_t)(p * 32) * K + kb + BK);
      }
      if (kb + 2 * BK < K) {  // global_prefetch_b8 two tiles ahead
        __builtin_prefetch(aptr + kb + 2 * BK, 0, 1);
        __builtin_prefetch(wptr + kb + 2 * BK, 0, 1);
      }
    }
    __syncthreads();
#pragma unroll
    for (int mi = 0; mi < 2; ++mi) {
      v16bf af = load_frag_a(As[cur], wr * 32 + mi * 16, 0, SA);
#pragma unroll
      for (int ni = 0; ni < 4; ++ni) {
        v16bf bf = load_frag_b(Bs[cur], wc * 64 + ni * 16, 0, SB);
        acc[mi][ni] = wmma_bf16(af, bf, acc[mi][ni]);
      }
    }
    if (has_next) {
#pragma unroll
      for (int p = 0; p < 4; ++p) {
        store_bf16x4(&As[cur ^ 1][(tr + p * 32) * SA + tk * 4], ra[p]);
        store_bf16x4(&Bs[cur ^ 1][(tr + p * 32) * SB + tk * 4], rb[p]);
      }
    }
    cur ^= 1;
  }

#pragma unroll
  for (int mi = 0; mi < 2; ++mi) {
    const int r0 = rowBase + wr * 32 + mi * 16;
#pragma unroll
    for (int ni = 0; ni < 4; ++ni) {
      const int c0 = colBase + wc * 64 + ni * 16 + ln;
      const float bv = bias[c0];
#pragma unroll
      for (int i = 0; i < 8; ++i) {
        Cmat[(size_t)(r0 + i + 8 * lh) * N + c0] = acc[mi][ni][i] + bv;
      }
    }
  }
}

// ---------------------------------------------------------------------------
// Fused feature-map + chunked causal linear-attention scan.
// One block per (b,h); 128 threads = 4 waves; chunk = 64 timesteps.
// Per chunk (all 64x64x64 WMMA matmuls):
//   Q' = relu(Q @ P), K' = relu(K @ P)
//   Acau = tril(Q' K'^T); norm = rowsum(Acau) + Q'.svec
//   ctx  = Q' @ S_prev + Acau @ V       -> y = ctx / (norm + 1e-6)
//   S   += K'^T @ V  (state held in accumulator registers per wave)
//   svec += colsum(K')
// ---------------------------------------------------------------------------
__global__ __launch_bounds__(128) void fastattn_scan_wmma(
    const float* __restrict__ qkv,   // [B,T,3C]
    const float* __restrict__ proj,  // [H,D,M]
    float* __restrict__ y)           // [B,T,C]
{
  extern __shared__ unsigned char smem_raw[];
  bf16_t* Pb  = (bf16_t*)smem_raw;          // [m][d]  B-layout for Q@P (n=m,k=d)
  bf16_t* Qa  = Pb  + 64 * ST;              // Q chunk [t][d], later Q' [t][m]
  bf16_t* Ka  = Qa  + 64 * ST;              // K chunk [t][d], later K' [s][m]
  bf16_t* Vb  = Ka  + 64 * ST;              // V chunk [d][t]  (B-layout, k = chunk time)
  bf16_t* KpT = Vb  + 64 * ST;              // K'^T [m][t]     (A for state update)
  bf16_t* Aa  = KpT + 64 * ST;              // masked scores [t][s]
  bf16_t* Sb  = Aa  + 64 * ST;              // S bf16 [d][m]   (B-layout for Q'@S)
  float*  svec  = (float*)(Sb + 64 * ST);   // [64] cumulative k' sum
  float*  normc = svec + 64;                // [64] per-chunk norm

  const int bh = blockIdx.x;
  const int b  = bh / HH;
  const int h  = bh % HH;
  const int tid  = threadIdx.x;
  const int wave = tid >> 5;
  const int lane = tid & 31;
  const int ln = lane & 15, lh = lane >> 4;

  const float* qbase = qkv + (size_t)b * TT * 3 * CC + h * DD;
  const float* kbase = qbase + CC;
  const float* vbase = qbase + 2 * CC;
  const float* pbase = proj + (size_t)h * DD * MM;
  float* ybase = y + (size_t)b * TT * CC + h * DD;

  // ---- init: P (transposed), Sb = 0, svec = 0 ----
  for (int idx = tid; idx < DD * MM; idx += 128) {
    int d2 = idx >> 6, m2 = idx & 63;
    Pb[m2 * ST + d2] = (bf16_t)pbase[d2 * MM + m2];
  }
  for (int idx = tid; idx < 64 * ST; idx += 128) Sb[idx] = (bf16_t)0.f;
  if (tid < 64) svec[tid] = 0.f;

  const v8f vzero = {0.f, 0.f, 0.f, 0.f, 0.f, 0.f, 0.f, 0.f};
  v8f sreg[4];   // this wave's 16xD strip of state S, persistent across chunks
#pragma unroll
  for (int ni = 0; ni < 4; ++ni) sreg[ni] = vzero;

  const int tk = tid & 15;   // float4 slot (d-dim)
  const int tr = tid >> 4;   // 0..7

  for (int c = 0; c < TT / LC; ++c) {
    __syncthreads();
    const int t0 = c * LC;

    // ---- phase 1: load Q,K,V chunk (batched loads, then f32 -> bf16 LDS) ----
    {
      float4 fq[8], fk[8], fv[8];
#pragma unroll
      for (int p = 0; p < 8; ++p) {
        const int r = tr + p * 8;
        const size_t rowoff = (size_t)(t0 + r) * 3 * CC + tk * 4;
        fq[p] = *(const float4*)(qbase + rowoff);
        fk[p] = *(const float4*)(kbase + rowoff);
        fv[p] = *(const float4*)(vbase + rowoff);
      }
      if (t0 + LC < TT) {  // prefetch next chunk's rows
        __builtin_prefetch(qbase + (size_t)(t0 + LC + tr * 8) * 3 * CC + tk * 4, 0, 1);
      }
#pragma unroll
      for (int p = 0; p < 8; ++p) {
        const int r = tr + p * 8;
        store_bf16x4(&Qa[r * ST + tk * 4], fq[p]);
        store_bf16x4(&Ka[r * ST + tk * 4], fk[p]);
        Vb[(tk * 4 + 0) * ST + r] = (bf16_t)fv[p].x;
        Vb[(tk * 4 + 1) * ST + r] = (bf16_t)fv[p].y;
        Vb[(tk * 4 + 2) * ST + r] = (bf16_t)fv[p].z;
        Vb[(tk * 4 + 3) * ST + r] = (bf16_t)fv[p].w;
      }
    }
    __syncthreads();

    // ---- phase 2: Q' = relu(Q@P), K' = relu(K@P) ----
    v8f qacc[4], kacc[4];
#pragma unroll
    for (int ni = 0; ni < 4; ++ni) { qacc[ni] = vzero; kacc[ni] = vzero; }
#pragma unroll
    for (int ks = 0; ks < 2; ++ks) {
      v16bf aq = load_frag_a(Qa, wave * 16, ks * 32, ST);
      v16bf ak = load_frag_a(Ka, wave * 16, ks * 32, ST);
#pragma unroll
      for (int ni = 0; ni < 4; ++ni) {
        v16bf bp = load_frag_b(Pb, ni * 16, ks * 32, ST);
        qacc[ni] = wmma_bf16(aq, bp, qacc[ni]);
        kacc[ni] = wmma_bf16(ak, bp, kacc[ni]);
      }
    }
    __syncthreads();  // all reads of Qa/Ka complete before overwrite
#pragma unroll
    for (int ni = 0; ni < 4; ++ni) {
#pragma unroll
      for (int i = 0; i < 8; ++i) {
        const int trow = wave * 16 + i + 8 * lh;   // chunk-local time
        const int cm   = ni * 16 + ln;             // feature m
        float qv = fmaxf(qacc[ni][i], 0.f);
        float kv = fmaxf(kacc[ni][i], 0.f);
        Qa[trow * ST + cm]  = (bf16_t)qv;          // Q' row-major
        Ka[trow * ST + cm]  = (bf16_t)kv;          // K' [s][m] (B-layout, n=s,k=m)
        KpT[cm * ST + trow] = (bf16_t)kv;          // K'^T row-major (A-layout)
      }
    }
    if (tid < 64) normc[tid] = 0.f;
    __syncthreads();

    // ---- phase 3: scores Acau = tril(Q' K'^T), rowsums + Q'.svec -> normc ----
    v8f aacc[4];
#pragma unroll
    for (int ni = 0; ni < 4; ++ni) aacc[ni] = vzero;
#pragma unroll
    for (int ks = 0; ks < 2; ++ks) {
      v16bf aq = load_frag_a(Qa, wave * 16, ks * 32, ST);
#pragma unroll
      for (int ni = 0; ni < 4; ++ni) {
        v16bf bk = load_frag_b(Ka, ni * 16, ks * 32, ST);
        aacc[ni] = wmma_bf16(aq, bk, aacc[ni]);
      }
    }
#pragma unroll
    for (int ni = 0; ni < 4; ++ni) {
#pragma unroll
      for (int i = 0; i < 8; ++i) {
        const int trow = wave * 16 + i + 8 * lh;
        const int scol = ni * 16 + ln;
        float av = (scol <= trow) ? aacc[ni][i] : 0.f;
        Aa[trow * ST + scol] = (bf16_t)av;
        atomicAdd(&normc[trow], av);
      }
    }
    if (tid < 64) {
      float dv = 0.f;
#pragma unroll 8
      for (int m2 = 0; m2 < MM; ++m2) dv += (float)Qa[tid * ST + m2] * svec[m2];
      atomicAdd(&normc[tid], dv);
    }
    __syncthreads();

    // ---- phase 4: ctx = Q' @ S_prev + Acau @ V ; write y ----
    v8f cacc[4];
#pragma unroll
    for (int ni = 0; ni < 4; ++ni) cacc[ni] = vzero;
#pragma unroll
    for (int ks = 0; ks < 2; ++ks) {
      v16bf aq = load_frag_a(Qa, wave * 16, ks * 32, ST);
#pragma unroll
      for (int ni = 0; ni < 4; ++ni) {
        v16bf bs = load_frag_b(Sb, ni * 16, ks * 32, ST);
        cacc[ni] = wmma_bf16(aq, bs, cacc[ni]);
      }
    }
#pragma unroll
    for (int ks = 0; ks < 2; ++ks) {
      v16bf aa = load_frag_a(Aa, wave * 16, ks * 32, ST);
#pragma unroll
      for (int ni = 0; ni < 4; ++ni) {
        v16bf bv = load_frag_b(Vb, ni * 16, ks * 32, ST);
        cacc[ni] = wmma_bf16(aa, bv, cacc[ni]);
      }
    }
#pragma unroll
    for (int ni = 0; ni < 4; ++ni) {
#pragma unroll
      for (int i = 0; i < 8; ++i) {
        const int trow = wave * 16 + i + 8 * lh;
        const int dcol = ni * 16 + ln;
        const float nz = normc[trow] + 1e-6f;
        ybase[(size_t)(t0 + trow) * CC + dcol] = cacc[ni][i] / nz;
      }
    }
    __syncthreads();  // Sb readers done before refresh

    // ---- phase 5: S += K'^T @ V (register state), refresh Sb, svec ----
#pragma unroll
    for (int ks = 0; ks < 2; ++ks) {
      v16bf am = load_frag_a(KpT, wave * 16, ks * 32, ST);
#pragma unroll
      for (int ni = 0; ni < 4; ++ni) {
        v16bf bv = load_frag_b(Vb, ni * 16, ks * 32, ST);
        sreg[ni] = wmma_bf16(am, bv, sreg[ni]);
      }
    }
#pragma unroll
    for (int ni = 0; ni < 4; ++ni) {
#pragma unroll
      for (int i = 0; i < 8; ++i) {
        const int mrow = wave * 16 + i + 8 * lh;
        const int dcol = ni * 16 + ln;
        Sb[dcol * ST + mrow] = (bf16_t)sreg[ni][i];   // S bf16, [d][m] B-layout
      }
    }
    if (tid < 64) {
      float s = 0.f;
#pragma unroll 8
      for (int t2 = 0; t2 < LC; ++t2) s += (float)KpT[tid * ST + t2];
      svec[tid] += s;
    }
  }
}

// ---------------------------------------------------------------------------
// Launch
// ---------------------------------------------------------------------------
extern "C" void kernel_launch(void* const* d_in, const int* in_sizes, int n_in,
                              void* d_out, int out_size, void* d_ws, size_t ws_size,
                              hipStream_t stream) {
  const float* x    = (const float*)d_in[0];   // [B,T,C]
  const float* Wqkv = (const float*)d_in[1];   // [3C,C]
  const float* bqkv = (const float*)d_in[2];   // [3C]
  const float* Wout = (const float*)d_in[3];   // [C,C]
  const float* bout = (const float*)d_in[4];   // [C]
  const float* proj = (const float*)d_in[5];   // [H,D,M]
  float* out = (float*)d_out;                  // [B,T,C]

  float* qkv = (float*)d_ws;                             // B*T*3C f32
  float* yws = qkv + (size_t)BB * TT * 3 * CC;           // B*T*C  f32

  const int Mrows = BB * TT;     // 4096

  // 1) qkv = x @ Wqkv^T + bqkv
  dim3 g1(Mrows / BM, (3 * CC) / BN);
  gemm_bias_wmma<<<g1, 256, 0, stream>>>(x, Wqkv, bqkv, qkv, Mrows, 3 * CC, CC);

  // 2) fused feature map + chunked causal linear-attention scan -> yws
  const size_t scan_smem = (size_t)7 * 64 * ST * sizeof(bf16_t) + 2 * 64 * sizeof(float);
  fastattn_scan_wmma<<<dim3(BB * HH), 128, scan_smem, stream>>>(qkv, proj, yws);

  // 3) out = yws @ Wout^T + bout
  dim3 g3(Mrows / BM, CC / BN);
  gemm_bias_wmma<<<g3, 256, 0, stream>>>(yws, Wout, bout, out, Mrows, CC, CC);
}
